// Decoder_44100724195717
// MI455X (gfx1250) — matrix operand order repaired and checked by reading, not compiled, not run on hardware
//
#include <hip/hip_runtime.h>
#include <stdint.h>

// ---------------------------------------------------------------------------
// Types for CDNA5 WMMA
// ---------------------------------------------------------------------------
typedef __attribute__((ext_vector_type(16))) __bf16 v16bf;
typedef __attribute__((ext_vector_type(8)))  float  v8f;

union Frag16 { uint4 u[2]; v16bf v; };

__device__ __forceinline__ unsigned short f2bf(float f) {
  unsigned int u = __float_as_uint(f);
  unsigned int r = u + 0x7FFFu + ((u >> 16) & 1u);  // round-to-nearest-even
  return (unsigned short)(r >> 16);
}

__device__ __forceinline__ float sigmoidf_(float x) {
  return 1.0f / (1.0f + __expf(-x));
}

// ---------------------------------------------------------------------------
// Generic bf16 WMMA GEMM:  C[M,N] = A[M,K](bf16) * B[K,N](bf16) (+ bias[N])
// Block: 256 threads = 8 waves. Block tile: 128(M) x 32(N). K chunk: 32.
// Requires M % 128 == 0, N % 32 == 0, K % 32 == 0 (true for all call sites).
// A fragment: ISA 16x32 bf16 layout -> two 128-bit global loads per lane.
// B chunk staged TRANSPOSED into LDS (Bst[n][k]) so each fragment is two
// 128-bit ds loads in exactly the ISA 32x16 B layout (no repacking VALU).
// ---------------------------------------------------------------------------
__global__ __launch_bounds__(256) void gemm_bf16_wmma(
    const unsigned short* __restrict__ A, int lda,
    const unsigned short* __restrict__ Bm, int ldb,
    const float* __restrict__ bias,
    float* __restrict__ C, long long ldc,
    int M, int N, int K)
{
  // transposed B chunk: [n][k], row stride 40 u16 = 80 B (16B aligned, 20 dw)
  __shared__ unsigned short Bst[32][40];

  const int tid  = threadIdx.x;
  const int wave = tid >> 5;
  const int lane = tid & 31;

  const int m0   = blockIdx.y * 128 + wave * 16;
  const int nblk = blockIdx.x * 32;

  v8f acc0 = {};
  v8f acc1 = {};

  const int mr    = m0 + (lane & 15);
  const int kbase = (lane < 16) ? 0 : 8;   // A: lanes 0-15 hold K 0..7/16..23; 16-31 hold 8..15/24..31
  const int koff  = (lane < 16) ? 0 : 8;   // B: lanes 0-15 hold K 0..7/16..23; 16-31 hold 8..15/24..31
  const int nlane = lane & 15;

  // per-thread staging coords: element idx = tid*4 -> (kk, nn..nn+3)
  const int kk = (tid * 4) >> 5;
  const int nn = (tid * 4) & 31;

  for (int k0 = 0; k0 < K; k0 += 32) {
    // ---- stage B[k0..k0+31, nblk..nblk+31] transposed into LDS ----
    {
      const unsigned short* src =
          Bm + (size_t)(k0 + kk) * (size_t)ldb + (size_t)(nblk + nn);
      union { uint2 d; unsigned short s[4]; } u;
      u.d = *(const uint2*)src;  // coalesced 8B load (row-major B)
#pragma unroll
      for (int i = 0; i < 4; ++i) Bst[nn + i][kk] = u.s[i];
      if (k0 + 32 < K)  // prefetch next chunk (global_prefetch_b8)
        __builtin_prefetch(src + (size_t)32 * (size_t)ldb, 0, 1);
    }
    // ---- A fragment: two 128-bit global loads per lane ----
    Frag16 af;
    {
      const unsigned short* ap = A + (size_t)mr * (size_t)lda + (size_t)(k0 + kbase);
      af.u[0] = *(const uint4*)ap;         // K = kbase .. kbase+7
      af.u[1] = *(const uint4*)(ap + 16);  // K = kbase+16 .. kbase+23
    }
    __syncthreads();

    // ---- B fragments: two 128-bit LDS loads each, already in ISA layout ----
    Frag16 b0, b1;
    b0.u[0] = *(const uint4*)&Bst[nlane][koff];
    b0.u[1] = *(const uint4*)&Bst[nlane][koff + 16];
    b1.u[0] = *(const uint4*)&Bst[nlane + 16][koff];
    b1.u[1] = *(const uint4*)&Bst[nlane + 16][koff + 16];

    acc0 = __builtin_amdgcn_wmma_f32_16x16x32_bf16(false, af.v, false, b0.v,
                                                   (short)0, acc0, false, false);
    acc1 = __builtin_amdgcn_wmma_f32_16x16x32_bf16(false, af.v, false, b1.v,
                                                   (short)0, acc1, false, false);
    __syncthreads();
  }

  // ---- epilogue: C/D layout: VGPR r -> (M = r + (lane<16?0:8), N = lane&15) ----
  const int rowb = m0 + ((lane < 16) ? 0 : 8);
  const int col0 = nblk + nlane;
  const int col1 = col0 + 16;
  const float bv0 = bias ? bias[col0] : 0.0f;
  const float bv1 = bias ? bias[col1] : 0.0f;
#pragma unroll
  for (int r = 0; r < 8; ++r) {
    int rr = rowb + r;
    if (rr < M) {
      C[(size_t)rr * (size_t)ldc + col0] = acc0[r] + bv0;
      C[(size_t)rr * (size_t)ldc + col1] = acc1[r] + bv1;
    }
  }
}

// ---------------------------------------------------------------------------
// fp32 -> bf16 conversion
// ---------------------------------------------------------------------------
__global__ void cvt_f32_bf16(const float* __restrict__ src,
                             unsigned short* __restrict__ dst, long long n)
{
  long long i = (long long)blockIdx.x * blockDim.x + threadIdx.x;
  if (i < n) dst[i] = f2bf(src[i]);
}

__global__ void bias_combine(const float* __restrict__ a,
                             const float* __restrict__ b,
                             float* __restrict__ o, int n)
{
  int i = blockIdx.x * blockDim.x + threadIdx.x;
  if (i < n) o[i] = a[i] + b[i];
}

// ---------------------------------------------------------------------------
// Bridge: h0/c0 = tanh(concat(final[0], final[1]) @ W + b)
// grid: (B, 2) ; y==0 -> h (written bf16 into Xg h-slot), y==1 -> c (fp32)
// ---------------------------------------------------------------------------
__global__ __launch_bounds__(256) void bridge_kernel(
    const float* __restrict__ fh, const float* __restrict__ fc,
    const float* __restrict__ Wbh, const float* __restrict__ bbh,
    const float* __restrict__ Wbc, const float* __restrict__ bbc,
    float* __restrict__ c32, unsigned short* __restrict__ Xg)
{
  const int b     = blockIdx.x;
  const int which = blockIdx.y;  // 0 = h, 1 = c
  const float* x  = (which ? fc : fh);
  const float* xf = x + (size_t)b * 1024;                       // final[0,b,:]
  const float* xb = x + (size_t)128 * 1024 + (size_t)b * 1024;  // final[1,b,:]
  const float* W  = which ? Wbc : Wbh;
  const float* bi = which ? bbc : bbh;

  for (int j = threadIdx.x; j < 1024; j += 256) {
    float acc = bi[j];
    for (int k = 0; k < 1024; ++k) acc += xf[k] * W[(size_t)k * 1024 + j];
    for (int k = 0; k < 1024; ++k) acc += xb[k] * W[(size_t)(1024 + k) * 1024 + j];
    acc = tanhf(acc);
    if (which) c32[b * 1024 + j] = acc;
    else       Xg[(size_t)b * 3584 + 2560 + j] = f2bf(acc);
  }
}

// ---------------------------------------------------------------------------
// Attention step: scores = v . tanh(q + proj_key); softmax; context.
// Packs bf16 activations into Xg ([x_t|ctx|h]) and Xp ([x_t|h_new|ctx]).
// One block per batch element b.
// ---------------------------------------------------------------------------
__global__ __launch_bounds__(256) void attn_kernel(
    const float* __restrict__ q,          // [B,1024]
    const float* __restrict__ proj_key,   // [B,128,1024]
    const float* __restrict__ vvec,       // [1024]
    const int*   __restrict__ src_mask,   // [B,1,128]
    const float* __restrict__ enc_hidden, // [B,128,2048]
    const unsigned short* __restrict__ tb,// [B,128,512] bf16
    unsigned short* __restrict__ Xg,      // [B,3584]
    unsigned short* __restrict__ Xp,      // [B,3584]
    int t)
{
  __shared__ float qs[1024];
  __shared__ float sc[128];
  __shared__ float al[128];

  const int b    = blockIdx.x;
  const int tid  = threadIdx.x;
  const int wave = tid >> 5;
  const int lane = tid & 31;

  for (int i = tid; i < 1024; i += 256) qs[i] = q[(size_t)b * 1024 + i];
  __syncthreads();

  // scores: each wave handles s = wave, wave+8, ...
  for (int s = wave; s < 128; s += 8) {
    const float* pk = proj_key + ((size_t)b * 128 + s) * 1024;
    float acc = 0.0f;
    for (int h = lane; h < 1024; h += 32)
      acc += vvec[h] * tanhf(qs[h] + pk[h]);
#pragma unroll
    for (int off = 16; off > 0; off >>= 1)
      acc += __shfl_down(acc, off, 32);
    if (lane == 0) {
      bool m = src_mask[b * 128 + s] != 0;
      sc[s] = m ? acc : -1e9f;
    }
  }
  __syncthreads();

  // softmax over 128 by wave 0
  if (wave == 0) {
    float x0 = sc[lane], x1 = sc[lane + 32], x2 = sc[lane + 64], x3 = sc[lane + 96];
    float mx = fmaxf(fmaxf(x0, x1), fmaxf(x2, x3));
#pragma unroll
    for (int off = 16; off > 0; off >>= 1) mx = fmaxf(mx, __shfl_xor(mx, off, 32));
    float e0 = __expf(x0 - mx), e1 = __expf(x1 - mx);
    float e2 = __expf(x2 - mx), e3 = __expf(x3 - mx);
    float sum = e0 + e1 + e2 + e3;
#pragma unroll
    for (int off = 16; off > 0; off >>= 1) sum += __shfl_xor(sum, off, 32);
    float inv = 1.0f / sum;
    al[lane] = e0 * inv; al[lane + 32] = e1 * inv;
    al[lane + 64] = e2 * inv; al[lane + 96] = e3 * inv;
  }
  __syncthreads();

  // context[d] = sum_s alpha[s] * enc_hidden[b,s,d]
  for (int d = tid; d < 2048; d += 256) {
    const float* eh = enc_hidden + (size_t)b * 128 * 2048 + d;
    float acc = 0.0f;
    for (int s = 0; s < 128; ++s) acc += al[s] * eh[(size_t)s * 2048];
    unsigned short bf = f2bf(acc);
    Xg[(size_t)b * 3584 + 512 + d]  = bf;  // gates input: cols 512..2559
    Xp[(size_t)b * 3584 + 1536 + d] = bf;  // pre   input: cols 1536..3583
  }

  // copy x_t (pre-converted bf16) into both panels
  const unsigned short* xt = tb + ((size_t)b * 128 + t) * 512;
  for (int i = tid; i < 512; i += 256) {
    unsigned short w = xt[i];
    Xg[(size_t)b * 3584 + i] = w;
    Xp[(size_t)b * 3584 + i] = w;
  }
}

// ---------------------------------------------------------------------------
// LSTM pointwise update. gates: [B,4096] (i|f|g|o). Updates c in place,
// writes h_new to decoder_states output and bf16 into Xg (h slot) / Xp (h slot).
// ---------------------------------------------------------------------------
__global__ __launch_bounds__(256) void lstm_kernel(
    const float* __restrict__ gates, float* __restrict__ c32,
    float* __restrict__ out_states,
    unsigned short* __restrict__ Xg, unsigned short* __restrict__ Xp, int t)
{
  int idx = blockIdx.x * blockDim.x + threadIdx.x;  // 0..131071
  int b = idx >> 10, j = idx & 1023;
  const float* g = gates + (size_t)b * 4096;
  float gi = g[j], gf = g[1024 + j], gg = g[2048 + j], go = g[3072 + j];
  float c  = c32[idx];
  float cn = sigmoidf_(gf) * c + sigmoidf_(gi) * tanhf(gg);
  float hn = sigmoidf_(go) * tanhf(cn);
  c32[idx] = cn;
  out_states[((size_t)b * 128 + t) * 1024 + j] = hn;
  unsigned short hb = f2bf(hn);
  Xg[(size_t)b * 3584 + 2560 + j] = hb;
  Xp[(size_t)b * 3584 + 512 + j]  = hb;
}

__global__ void final_copy(const float* __restrict__ out_states,
                           const float* __restrict__ c32,
                           float* __restrict__ hT, float* __restrict__ cT)
{
  int idx = blockIdx.x * blockDim.x + threadIdx.x;  // 131072
  int b = idx >> 10, j = idx & 1023;
  hT[idx] = out_states[((size_t)b * 128 + 127) * 1024 + j];
  cT[idx] = c32[idx];
}

// ---------------------------------------------------------------------------
// Host launcher
// ---------------------------------------------------------------------------
extern "C" void kernel_launch(void* const* d_in, const int* in_sizes, int n_in,
                              void* d_out, int out_size, void* d_ws, size_t ws_size,
                              hipStream_t stream)
{
  (void)in_sizes; (void)n_in; (void)out_size; (void)ws_size;

  const float* trg_embed  = (const float*)d_in[0];
  const float* enc_hidden = (const float*)d_in[1];
  const float* enc_fh     = (const float*)d_in[2];
  const float* enc_fc     = (const float*)d_in[3];
  const int*   src_mask   = (const int*)  d_in[4];
  const float* Wk         = (const float*)d_in[5];
  const float* Wq         = (const float*)d_in[6];
  const float* vvec       = (const float*)d_in[7];
  const float* W_ih       = (const float*)d_in[8];
  const float* W_hh       = (const float*)d_in[9];
  const float* b_ih       = (const float*)d_in[10];
  const float* b_hh       = (const float*)d_in[11];
  const float* Wbh        = (const float*)d_in[12];
  const float* bbh        = (const float*)d_in[13];
  const float* Wbc        = (const float*)d_in[14];
  const float* bbc        = (const float*)d_in[15];
  const float* Wpre       = (const float*)d_in[16];

  const int B = 128, S = 128, T = 128, E = 512, H = 1024;

  // workspace carve-out
  char* ws = (char*)d_ws;
  size_t off = 0;
  auto alloc = [&](size_t bytes) -> void* {
    void* p = ws + off;
    off += (bytes + 255) & ~(size_t)255;
    return p;
  };
  unsigned short* tb     = (unsigned short*)alloc((size_t)B * T * E * 2);        // trg_embed bf16
  unsigned short* ehb    = (unsigned short*)alloc((size_t)B * S * 2 * H * 2);    // enc_hidden bf16
  unsigned short* Wk_b   = (unsigned short*)alloc((size_t)2 * H * H * 2);
  unsigned short* Wq_b   = (unsigned short*)alloc((size_t)H * H * 2);
  unsigned short* Wg_b   = (unsigned short*)alloc((size_t)3584 * 4096 * 2);      // [W_ih; W_hh]
  unsigned short* Wpre_b = (unsigned short*)alloc((size_t)3584 * 1024 * 2);
  float* pk    = (float*)alloc((size_t)B * S * H * 4);                            // proj_key
  float* qbuf  = (float*)alloc((size_t)B * H * 4);
  float* gates = (float*)alloc((size_t)B * 4 * H * 4);
  float* c32   = (float*)alloc((size_t)B * H * 4);
  unsigned short* Xg = (unsigned short*)alloc((size_t)B * 3584 * 2);
  unsigned short* Xp = (unsigned short*)alloc((size_t)B * 3584 * 2);
  float* bgate = (float*)alloc((size_t)4 * H * 4);

  float* out        = (float*)d_out;
  float* out_states = out;                                   // [B,T,H]
  float* out_hT     = out + (size_t)B * T * H;               // [B,H]
  float* out_cT     = out_hT + (size_t)B * H;                // [B,H]
  float* out_pres   = out_cT + (size_t)B * H;                // [B,T,H]

  auto cvt = [&](const float* s, unsigned short* d, long long n) {
    int blocks = (int)((n + 255) / 256);
    cvt_f32_bf16<<<blocks, 256, 0, stream>>>(s, d, n);
  };

  // ---- one-time conversions ----
  cvt(trg_embed,  tb,   (long long)B * T * E);
  cvt(enc_hidden, ehb,  (long long)B * S * 2 * H);
  cvt(Wk,         Wk_b, (long long)2 * H * H);
  cvt(Wq,         Wq_b, (long long)H * H);
  cvt(W_ih,       Wg_b, (long long)2560 * 4096);
  cvt(W_hh,       Wg_b + (size_t)2560 * 4096, (long long)H * 4 * H);
  cvt(Wpre,       Wpre_b, (long long)3584 * 1024);
  bias_combine<<<16, 256, 0, stream>>>(b_ih, b_hh, bgate, 4 * H);

  // ---- bridges: h0 (bf16 into Xg h-slot), c0 (fp32) ----
  bridge_kernel<<<dim3(B, 2), 256, 0, stream>>>(enc_fh, enc_fc, Wbh, bbh, Wbc, bbc, c32, Xg);

  // ---- proj_key = enc_hidden @ Wk : [16384,2048] x [2048,1024] ----
  gemm_bf16_wmma<<<dim3(1024 / 32, 16384 / 128), 256, 0, stream>>>(
      ehb, 2 * H, Wk_b, H, nullptr, pk, H, B * S, H, 2 * H);

  // ---- recurrent loop ----
  for (int t = 0; t < T; ++t) {
    // q = h @ Wq  (A = h-slot of Xg, lda=3584)
    gemm_bf16_wmma<<<dim3(H / 32, 1), 256, 0, stream>>>(
        Xg + 2560, 3584, Wq_b, H, nullptr, qbuf, H, B, H, H);

    attn_kernel<<<B, 256, 0, stream>>>(qbuf, pk, vvec, src_mask, enc_hidden,
                                       tb, Xg, Xp, t);

    // gates = [x_t|ctx|h] @ [W_ih;W_hh] + (b_ih+b_hh)
    gemm_bf16_wmma<<<dim3(4 * H / 32, 1), 256, 0, stream>>>(
        Xg, 3584, Wg_b, 4 * H, bgate, gates, 4 * H, B, 4 * H, 3584);

    lstm_kernel<<<(B * H) / 256, 256, 0, stream>>>(gates, c32, out_states, Xg, Xp, t);

    // pre = [x_t|h_new|ctx] @ Wpre -> directly into output slice [:, t, :]
    gemm_bf16_wmma<<<dim3(H / 32, 1), 256, 0, stream>>>(
        Xp, 3584, Wpre_b, H, nullptr, out_pres + (size_t)t * H, (long long)T * H,
        B, H, 3584);
  }

  final_copy<<<(B * H) / 256, 256, 0, stream>>>(out_states, c32, out_hT, out_cT);
}
